// TransducerDecoder_64415919506130
// MI455X (gfx1250) — matrix operand order repaired and checked by reading, not compiled, not run on hardware
//
#include <hip/hip_runtime.h>
#include <math.h>

// ---------------- problem constants ----------------
#define BSZ    2
#define TLEN   200
#define ULEN   50
#define DMODEL 256
#define INNERD 2048
#define VOCAB  5000
#define NPAD   5120          // VOCAB padded to 320 tiles of 16
#define TT16   13            // ceil(TLEN/16): t-tiles of 16
#define UB4    13            // ceil(ULEN/4):  u-blocks of 4
#define NTILES16 1300        // BSZ * ULEN * TT16  (A fragment tiles)
#define TILE_HALVES 32768    // 16 rows x 2048 k (halves) per 16-row tile / n-tile

typedef __attribute__((ext_vector_type(16))) _Float16 v16h;
typedef __attribute__((ext_vector_type(8)))  _Float16 v8h;
typedef __attribute__((ext_vector_type(8)))  float    v8f;

__device__ __forceinline__ float sigmoidf_(float x) { return 1.0f / (1.0f + __expf(-x)); }

// ---------------------------------------------------------------
// K0: pack W_out (fp32 [V, INNER]) into f16 WMMA B-fragment layout:
//     Bp[nt*32768 + kbi*512 + lane*16 + h] = (f16) W_out[nt*16+n][k]  (zero pad n>=V)
// B fragment (32x16): lane = g*16+n, halves h -> K as in the A table. One contiguous
// 32B load per lane per k-step in the GEMM.
// ---------------------------------------------------------------
__global__ __launch_bounds__(256)
void pack_wout_k(const float* __restrict__ W_out, _Float16* __restrict__ Bp) {
    int idx = blockIdx.x * 256 + threadIdx.x;      // over NPAD*INNERD
    if (idx >= NPAD * INNERD) return;
    int nrow = idx / INNERD;
    int k    = idx - nrow * INNERD;
    float v = 0.0f;
    if (nrow < VOCAB) v = W_out[(size_t)nrow * INNERD + k];
    Bp[(size_t)(nrow >> 4) * TILE_HALVES + (size_t)(k >> 5) * 512 + (size_t)((((k >> 3) & 1) * 16 + (nrow & 15)) * 16) + (k & 7) + ((k >> 4) & 1) * 8] = (_Float16)v;
}

// ---------------------------------------------------------------
// K1: embedding gather. grid = B*U blocks, 256 threads (= DMODEL)
// ---------------------------------------------------------------
__global__ __launch_bounds__(256)
void embed_k(const int* __restrict__ ids, const float* __restrict__ emb,
             float* __restrict__ x) {
    int row = blockIdx.x;
    int tid = threadIdx.x;
    int id = ids[row];
    x[(size_t)row * DMODEL + tid] = emb[(size_t)id * DMODEL + tid];
}

// ---------------------------------------------------------------
// Generic small row-GEMM: Y[row, j] = bias[j] + sum_k W[j,k] * X[row,k]
// K = DMODEL (=256) held in LDS. grid = #rows, 256 threads.
// ---------------------------------------------------------------
template <int NOUT>
__global__ __launch_bounds__(256)
void rowgemm_k(const float* __restrict__ X, const float* __restrict__ W,
               const float* __restrict__ bias, float* __restrict__ Y) {
    __shared__ float xr[DMODEL];
    int row = blockIdx.x;
    int tid = threadIdx.x;
    if (tid < DMODEL) xr[tid] = X[(size_t)row * DMODEL + tid];
    __syncthreads();
    for (int j = tid; j < NOUT; j += 256) {
        const float* w = W + (size_t)j * DMODEL;
        float acc = bias[j];
        #pragma unroll 8
        for (int k = 0; k < DMODEL; ++k) acc += w[k] * xr[k];
        Y[(size_t)row * NOUT + j] = acc;
    }
}

// ---------------------------------------------------------------
// K2: LSTM recurrence (torch gate order i,f,g,o). One block per batch,
// 1024 threads = one gate output each; h, gates in LDS, c in registers.
// xW already holds x@W_ih.T + b_ih for every (b,t).
// ---------------------------------------------------------------
__global__ __launch_bounds__(1024)
void lstm_k(const float* __restrict__ xW, const float* __restrict__ W_hh,
            const float* __restrict__ b_hh, float* __restrict__ hs) {
    __shared__ float hsh[DMODEL];
    __shared__ float gates[4 * DMODEL];
    int b = blockIdx.x;
    int tid = threadIdx.x;
    float c = 0.0f;
    if (tid < DMODEL) hsh[tid] = 0.0f;
    __syncthreads();
    const float bb = b_hh[tid];
    const float* w = W_hh + (size_t)tid * DMODEL;
    for (int t = 0; t < ULEN; ++t) {
        float acc = bb + xW[((size_t)b * ULEN + t) * (4 * DMODEL) + tid];
        #pragma unroll 8
        for (int k = 0; k < DMODEL; ++k) acc += w[k] * hsh[k];
        gates[tid] = acc;
        __syncthreads();
        if (tid < DMODEL) {
            float ig = gates[tid];
            float fg = gates[DMODEL + tid];
            float gg = gates[2 * DMODEL + tid];
            float og = gates[3 * DMODEL + tid];
            c = sigmoidf_(fg) * c + sigmoidf_(ig) * tanhf(gg);
            float hn = sigmoidf_(og) * tanhf(c);
            hsh[tid] = hn;
            hs[((size_t)b * ULEN + t) * DMODEL + tid] = hn;
        }
        __syncthreads();
    }
}

// ---------------------------------------------------------------
// K3: projection + LayerNorm fused. grid = B*U, 256 threads (= DMODEL).
// ---------------------------------------------------------------
__global__ __launch_bounds__(256)
void projln_k(const float* __restrict__ hs, const float* __restrict__ W_proj,
              const float* __restrict__ b_proj, const float* __restrict__ ln_g,
              const float* __restrict__ ln_b, float* __restrict__ lnout) {
    __shared__ float hrow[DMODEL];
    __shared__ float red[256];
    int row = blockIdx.x;
    int tid = threadIdx.x;
    hrow[tid] = hs[(size_t)row * DMODEL + tid];
    __syncthreads();
    const float* w = W_proj + (size_t)tid * DMODEL;
    float y = b_proj[tid];
    #pragma unroll 8
    for (int k = 0; k < DMODEL; ++k) y += w[k] * hrow[k];
    red[tid] = y;
    __syncthreads();
    for (int s = 128; s > 0; s >>= 1) {
        if (tid < s) red[tid] += red[tid + s];
        __syncthreads();
    }
    float mu = red[0] * (1.0f / DMODEL);
    __syncthreads();
    float d = y - mu;
    red[tid] = d * d;
    __syncthreads();
    for (int s = 128; s > 0; s >>= 1) {
        if (tid < s) red[tid] += red[tid + s];
        __syncthreads();
    }
    float var = red[0] * (1.0f / DMODEL);
    float r = rsqrtf(var + 1e-5f);
    lnout[(size_t)row * DMODEL + tid] = d * r * ln_g[tid] + ln_b[tid];
}

// ---------------------------------------------------------------
// K4: build the f16 joint tensor ONCE in WMMA A-fragment-linear layout.
// A fragment (16x32, ISA table): lane = g*16+m holds
//   halves 0..7  -> K = kb + g*8 + h
//   halves 8..15 -> K = kb + 16 + g*8 + (h-8)
// Af[tile][kbi][lane][h] stored linearly => per (tile,kbi) each lane's
// v16h is one contiguous 32B chunk at Af + tile*32768 + kbi*512 + lane*16.
// One thread produces one 8-half chunk (8 contiguous k) => coalesced
// reads of enc/dec and coalesced 16B writes. tile id = (b*ULEN+u)*TT16+tt,
// rows = 16 consecutive t (clamped at TLEN).
// Total tanh = 20000*2048 = 41M (computed exactly once).
// ---------------------------------------------------------------
__global__ __launch_bounds__(256)
void joint_build_k(const float* __restrict__ enc, const float* __restrict__ dec,
                   _Float16* __restrict__ Af) {
    int cid  = blockIdx.x * 256 + threadIdx.x;   // NTILES16*4096 chunks
    int c    = cid & 4095;                       // chunk within tile
    int tile = cid >> 12;
    int kbi  = c >> 6;                           // k-block of 32
    int lane = (c >> 1) & 31;
    int hgrp = c & 1;                            // halves 0..7 / 8..15
    int tt = tile % TT16;
    int q  = tile / TT16;
    int u  = q % ULEN;
    int b  = q / ULEN;
    int m  = lane & 15;
    int t  = tt * 16 + m;
    if (t >= TLEN) t = TLEN - 1;                 // pad rows = dup, never stored
    int k0 = kbi * 32 + hgrp * 16 + (lane >> 4) * 8;
    const float* e = enc + ((size_t)b * TLEN + t) * INNERD + k0;
    const float* d = dec + ((size_t)b * ULEN + u) * INNERD + k0;
    v8h o;
    #pragma unroll
    for (int i = 0; i < 8; ++i) o[i] = (_Float16)tanhf(e[i] + d[i]);
    *(v8h*)(Af + (size_t)cid * 8) = o;
}

// ---------------------------------------------------------------
// K5: the big GEMM, pure register streaming (no LDS, no barriers).
// Block = (b, ub, tt): 64 M-rows (4 u-subtiles x 16 t-rows) x 256 N cols
// (8 waves x 2 n-tiles). 64 K-steps, ping-pong double-buffered so A/B
// loads for step k+1 are in flight across the 8 WMMAs of step k.
// acc: 4x2 v8f = 64 VGPRs; bufs: (4+2)x2 v16h = 96 VGPRs.
// C/D (16x16 f32): VGPR r -> M = g*8 + r, N = lane&15.
// ---------------------------------------------------------------
__global__ __launch_bounds__(256)
void joint_gemm_k(const _Float16* __restrict__ Af, const _Float16* __restrict__ Bp,
                  const float* __restrict__ b_out, float* __restrict__ out) {
    const int bx = blockIdx.x;                   // [0, BSZ*UB4*TT16)
    const int tt = bx % TT16;
    const int q  = bx / TT16;
    const int ub = q % UB4;
    const int b  = q / UB4;
    const int z  = blockIdx.y;                   // [0, 20): N macro-block of 256
    const int tid    = threadIdx.x;
    const int wave   = tid >> 5;
    const int lane   = tid & 31;
    const int laneHi = lane >> 4;                // g
    const int lane15 = lane & 15;                // n (for C/D)

    const _Float16* ap[4];
    #pragma unroll
    for (int s = 0; s < 4; ++s) {
        int u = ub * 4 + s;
        if (u >= ULEN) u = ULEN - 1;             // dup tile; stores guarded
        int tile = (b * ULEN + u) * TT16 + tt;
        ap[s] = Af + (size_t)tile * TILE_HALVES + (size_t)lane * 16;
    }
    const _Float16* bp[2];
    #pragma unroll
    for (int j = 0; j < 2; ++j) {
        int nt = (z * 8 + wave) * 2 + j;         // bijective over [0,320)
        bp[j] = Bp + (size_t)nt * TILE_HALVES + (size_t)lane * 16;
    }

    v8f acc[4][2] = {};
    v16h a0[4], a1[4], b0[2], b1[2];

    #pragma unroll
    for (int s = 0; s < 4; ++s) a0[s] = *(const v16h*)(ap[s]);
    #pragma unroll
    for (int j = 0; j < 2; ++j) b0[j] = *(const v16h*)(bp[j]);

    for (int kbi = 0; kbi < 64; kbi += 2) {
        const size_t off1 = (size_t)(kbi + 1) * 512;
        #pragma unroll
        for (int s = 0; s < 4; ++s) a1[s] = *(const v16h*)(ap[s] + off1);
        #pragma unroll
        for (int j = 0; j < 2; ++j) b1[j] = *(const v16h*)(bp[j] + off1);
        #pragma unroll
        for (int s = 0; s < 4; ++s)
            #pragma unroll
            for (int j = 0; j < 2; ++j)
                acc[s][j] = __builtin_amdgcn_wmma_f32_16x16x32_f16(
                    false, a0[s], false, b0[j], (short)0, acc[s][j], false, false);

        const int k2 = (kbi + 2 < 64) ? (kbi + 2) : 63;  // last reload unused
        const size_t off2 = (size_t)k2 * 512;
        #pragma unroll
        for (int s = 0; s < 4; ++s) a0[s] = *(const v16h*)(ap[s] + off2);
        #pragma unroll
        for (int j = 0; j < 2; ++j) b0[j] = *(const v16h*)(bp[j] + off2);
        #pragma unroll
        for (int s = 0; s < 4; ++s)
            #pragma unroll
            for (int j = 0; j < 2; ++j)
                acc[s][j] = __builtin_amdgcn_wmma_f32_16x16x32_f16(
                    false, a1[s], false, b1[j], (short)0, acc[s][j], false, false);
    }

    // store D + b_out (each output element written exactly once grid-wide)
    #pragma unroll
    for (int s = 0; s < 4; ++s) {
        int u = ub * 4 + s;
        if (u < ULEN) {
            #pragma unroll
            for (int j = 0; j < 2; ++j) {
                int nt = (z * 8 + wave) * 2 + j;
                int n = nt * 16 + lane15;
                if (n < VOCAB) {
                    float bo = b_out[n];
                    #pragma unroll
                    for (int r = 0; r < 8; ++r) {
                        int t = tt * 16 + laneHi * 8 + r;
                        if (t < TLEN)
                            out[(((size_t)b * TLEN + t) * ULEN + u) * VOCAB + n] =
                                acc[s][j][r] + bo;
                    }
                }
            }
        }
    }
}

// ---------------------------------------------------------------
// K6: in-place log_softmax over last dim. grid = B*T*U, 256 threads.
// ---------------------------------------------------------------
__global__ __launch_bounds__(256)
void logsoftmax_k(float* __restrict__ out) {
    __shared__ float red[256];
    float* p = out + (size_t)blockIdx.x * VOCAB;
    int tid = threadIdx.x;

    float mx = -__builtin_inff();
    for (int i = tid; i < VOCAB; i += 256) mx = fmaxf(mx, p[i]);
    red[tid] = mx;
    __syncthreads();
    for (int s = 128; s > 0; s >>= 1) {
        if (tid < s) red[tid] = fmaxf(red[tid], red[tid + s]);
        __syncthreads();
    }
    mx = red[0];
    __syncthreads();

    float sum = 0.0f;
    for (int i = tid; i < VOCAB; i += 256) sum += __expf(p[i] - mx);
    red[tid] = sum;
    __syncthreads();
    for (int s = 128; s > 0; s >>= 1) {
        if (tid < s) red[tid] += red[tid + s];
        __syncthreads();
    }
    float lse = mx + __logf(red[0]);
    __syncthreads();
    for (int i = tid; i < VOCAB; i += 256) p[i] = p[i] - lse;
}

// ---------------------------------------------------------------
// Launch
// ---------------------------------------------------------------
extern "C" void kernel_launch(void* const* d_in, const int* in_sizes, int n_in,
                              void* d_out, int out_size, void* d_ws, size_t ws_size,
                              hipStream_t stream) {
    (void)in_sizes; (void)n_in; (void)out_size; (void)ws_size;

    const float* memory = (const float*)d_in[0];
    const int*   ys     = (const int*)d_in[1];
    /* d_in[2] = ys_in_lens : unused by reference output */
    const float* emb    = (const float*)d_in[3];
    const float* W_ih   = (const float*)d_in[4];
    const float* b_ih   = (const float*)d_in[5];
    const float* W_hh   = (const float*)d_in[6];
    const float* b_hh   = (const float*)d_in[7];
    const float* W_proj = (const float*)d_in[8];
    const float* b_proj = (const float*)d_in[9];
    const float* ln_g   = (const float*)d_in[10];
    const float* ln_b   = (const float*)d_in[11];
    const float* W_enc  = (const float*)d_in[12];
    const float* b_enc  = (const float*)d_in[13];
    const float* W_prd  = (const float*)d_in[14];
    const float* b_prd  = (const float*)d_in[15];
    const float* W_out  = (const float*)d_in[16];
    const float* b_out  = (const float*)d_in[17];
    float* out = (float*)d_out;

    // workspace layout (aligned offsets), total ~106 MB
    char* ws = (char*)d_ws;
    float*    x    = (float*)(ws + 0);          //  100*256*4   = 102400
    float*    xW   = (float*)(ws + 102400);     //  100*1024*4  = 409600
    float*    hsb  = (float*)(ws + 512000);     //  100*256*4   = 102400
    float*    ln   = (float*)(ws + 614400);     //  100*256*4   = 102400
    float*    decb = (float*)(ws + 716800);     //  100*2048*4  = 819200
    float*    encb = (float*)(ws + 1536000);    //  400*2048*4  = 3276800
    _Float16* Bp   = (_Float16*)(ws + 4812800); // 5120*2048*2  = 20971520
    _Float16* Af   = (_Float16*)(ws + 25784320);// 1300*32768*2 = 85196800

    // pack W_out (one-time per launch; W_out f16 stays L2-resident)
    pack_wout_k<<<(NPAD * INNERD + 255) / 256, 256, 0, stream>>>(W_out, Bp);

    // prediction network
    embed_k<<<BSZ * ULEN, 256, 0, stream>>>(ys, emb, x);
    rowgemm_k<4 * DMODEL><<<BSZ * ULEN, 256, 0, stream>>>(x, W_ih, b_ih, xW);
    lstm_k<<<BSZ, 1024, 0, stream>>>(xW, W_hh, b_hh, hsb);
    projln_k<<<BSZ * ULEN, 256, 0, stream>>>(hsb, W_proj, b_proj, ln_g, ln_b, ln);

    // joint-net pre-linears
    rowgemm_k<INNERD><<<BSZ * ULEN, 256, 0, stream>>>(ln, W_prd, b_prd, decb);
    rowgemm_k<INNERD><<<BSZ * TLEN, 256, 0, stream>>>(memory, W_enc, b_enc, encb);

    // f16 joint tensor in WMMA fragment layout (82 MB, L2-resident)
    joint_build_k<<<(NTILES16 * 4096) / 256, 256, 0, stream>>>(encb, decb, Af);

    // big WMMA GEMM: logits -> d_out
    dim3 grid5(BSZ * UB4 * TT16, NPAD / 256);   // 338 x 20
    joint_gemm_k<<<grid5, 256, 0, stream>>>(Af, Bp, b_out, out);

    // in-place log_softmax
    logsoftmax_k<<<BSZ * TLEN * ULEN, 256, 0, stream>>>(out);
}
